// DecompMultiTransform_1958505087671
// MI455X (gfx1250) — matrix operand order repaired and checked by reading, not compiled, not run on hardware
//
#include <hip/hip_runtime.h>

#define IN_DIM  256
#define OUT_DIM 256
#define NUM_T   64
#define NUM_B   16
#define NN      4096

typedef __attribute__((ext_vector_type(16))) _Float16 v16h;
typedef __attribute__((ext_vector_type(8)))  _Float16 v8h;
typedef __attribute__((ext_vector_type(8)))  float    v8f;

// ---------------------------------------------------------------------------
// Phase 0a: Wt[t][o][i] = sum_b w_comp[t][b] * weight[b][i*OUT_DIM + o]  (f16)
// Pre-transposed per-type weights so WMMA B fragments are contiguous-K loads.
// grid = (16 t-groups of 4, 4 i-slices of 64), block = 256 (thread == o).
// weight rows are read fully coalesced; each sweep is reused across 4 types.
// ---------------------------------------------------------------------------
__global__ void __launch_bounds__(256) build_wt(const float* __restrict__ weight,
                                                const float* __restrict__ w_comp,
                                                _Float16* __restrict__ Wt) {
  const int o  = threadIdx.x;          // 0..255
  const int t0 = blockIdx.x * 4;       // 4 types per block
  const int i0 = blockIdx.y * 64;      // 64 input-dims per block

  float wc[4][NUM_B];
  for (int tt = 0; tt < 4; ++tt)
    for (int b = 0; b < NUM_B; ++b)
      wc[tt][b] = w_comp[(t0 + tt) * NUM_B + b];

  for (int ic = 0; ic < 64; ic += 16) {
    float acc[4][16];
    for (int tt = 0; tt < 4; ++tt)
      for (int ii = 0; ii < 16; ++ii) acc[tt][ii] = 0.0f;

    for (int ii = 0; ii < 16; ++ii) {
      const int i = i0 + ic + ii;
      for (int b = 0; b < NUM_B; ++b) {
        const float w = weight[(size_t)b * (IN_DIM * OUT_DIM) + (size_t)i * OUT_DIM + o];
        for (int tt = 0; tt < 4; ++tt) acc[tt][ii] += wc[tt][b] * w;
      }
    }

    for (int tt = 0; tt < 4; ++tt) {
      v8h lo, hi;
      for (int ii = 0; ii < 8; ++ii) {
        lo[ii] = (_Float16)acc[tt][ii];
        hi[ii] = (_Float16)acc[tt][ii + 8];
      }
      _Float16* dst = Wt + ((size_t)(t0 + tt) * OUT_DIM + o) * IN_DIM + i0 + ic;
      *(v8h*)(dst)     = lo;   // 16B contiguous per thread
      *(v8h*)(dst + 8) = hi;
    }
  }
}

// ---------------------------------------------------------------------------
// Phase 0b: x (f32) -> x16 (f16), fully coalesced. grid = NN*IN_DIM/256.
// ---------------------------------------------------------------------------
__global__ void __launch_bounds__(256) cvt_x(const float* __restrict__ x,
                                             _Float16* __restrict__ x16) {
  const int idx = blockIdx.x * 256 + threadIdx.x;
  x16[idx] = (_Float16)x[idx];
}

// ---------------------------------------------------------------------------
// Phase 1: segment matmul. grid = (NUM_T types, 4 n-quarters), block = 256
// (8 wave32). Gather rows of this type into LDS, then WMMA-tile:
// M in 16-row chunks, N: 4 tiles of 16 per block (one per wave, 2 M-stripes),
// K = 256 in 8 steps of 32 with v_wmma_f32_16x16x32_f16.
// B fragments (8 x v16h) are loaded once per block and stay in VGPRs.
// A scheduling barrier pins all 16 A-fragment b128 loads of a tile BEFORE the
// 8 WMMAs, so the loads go out as one clause and the WMMAs drain them with
// partial loadcnt waits instead of a full load-wait per K-step.
// ---------------------------------------------------------------------------
__global__ void __launch_bounds__(256) rgcn_gemm(const _Float16* __restrict__ x16,
                                                 const int* __restrict__ xtype,
                                                 const _Float16* __restrict__ Wt,
                                                 float* __restrict__ out) {
  __shared__ int s_rows[NN];
  __shared__ int s_cnt;

  const int t = blockIdx.x;   // relation type
  const int q = blockIdx.y;   // quarter of the 16 n-tiles

  if (threadIdx.x == 0) s_cnt = 0;
  __syncthreads();
  for (int n = threadIdx.x; n < NN; n += blockDim.x) {
    if (xtype[n] == t) {
      int p = atomicAdd(&s_cnt, 1);
      s_rows[p] = n;
    }
  }
  __syncthreads();

  const int cnt = s_cnt;
  const int lane   = threadIdx.x & 31;
  const int wave   = threadIdx.x >> 5;
  const int lm     = lane & 15;
  const int lohalf = (lane < 16);

  const int ntile  = q * 4 + (wave & 3);   // 0..15
  const int n0     = ntile * 16;
  const int mstart = wave >> 2;            // 0 or 1; waves stride M-tiles by 2

  // B fragments: lane holds column (n0+lm); K contiguous, base 0 (lanes 0-15)
  // or 16 (lanes 16-31).  (Contiguous-K order, same as the A fragment.)  Loaded once, kept in
  // VGPRs for the whole M-loop (64 VGPRs).
  const _Float16* wcol =
      Wt + ((size_t)t * OUT_DIM + n0 + lm) * IN_DIM + (lohalf ? 0 : 16);
  v16h bfrag[8];
#pragma unroll
  for (int s = 0; s < 8; ++s) {
    v8h blo = *(const v8h*)(wcol + s * 32);      // K s*32+{0..7}  (or +16..23)
    v8h bhi = *(const v8h*)(wcol + s * 32 + 8);  // K s*32+{8..15} (or +24..31)
    bfrag[s] = __builtin_shufflevector(blo, bhi,
                  0,1,2,3,4,5,6,7,8,9,10,11,12,13,14,15);
  }

  // A fragment: lane holds row M=lm; K = {0..15} or {16..31}.
  const int aoff = lohalf ? 0 : 16;

  const int mtiles = (cnt + 15) >> 4;
  for (int mt = mstart; mt < mtiles; mt += 2) {
    const int mbase = mt * 16;
    const int mrow  = mbase + lm;
    const int srow  = s_rows[mrow < cnt ? mrow : cnt - 1];  // clamp tail (select)
    const _Float16* arow = x16 + (size_t)srow * IN_DIM + aoff;

    // Issue all 16 A-fragment loads for this tile up front (one clause).
    v8h afrag[16];
#pragma unroll
    for (int s = 0; s < 8; ++s) {
      afrag[2 * s]     = *(const v8h*)(arow + s * 32);       // K s*32+{0..7}/{16..23}
      afrag[2 * s + 1] = *(const v8h*)(arow + s * 32 + 8);   // K s*32+{8..15}/{24..31}
    }

    // Hard scheduling fence: nothing crosses.  Keeps all 16 loads in flight
    // before the first WMMA instead of load/wait/wmma lockstep.
    __builtin_amdgcn_sched_barrier(0);

    v8f c = {};
#pragma unroll
    for (int s = 0; s < 8; ++s) {
      v16h a = __builtin_shufflevector(afrag[2 * s], afrag[2 * s + 1],
                  0,1,2,3,4,5,6,7,8,9,10,11,12,13,14,15);
      c = __builtin_amdgcn_wmma_f32_16x16x32_f16(false, a, false, bfrag[s],
                                                 (short)0, c, false, false);
    }

    // D layout: VGPR r -> (M=r, N=lane) lanes 0-15; (M=8+r, N=lane-16) lanes 16-31.
    const int mhalf = lohalf ? 0 : 8;
    const int ncol  = n0 + lm;
#pragma unroll
    for (int r = 0; r < 8; ++r) {
      const int m = mbase + mhalf + r;
      if (m < cnt) out[(size_t)s_rows[m] * OUT_DIM + ncol] = c[r];
    }
  }
}

// ---------------------------------------------------------------------------
extern "C" void kernel_launch(void* const* d_in, const int* in_sizes, int n_in,
                              void* d_out, int out_size, void* d_ws, size_t ws_size,
                              hipStream_t stream) {
  const float* x      = (const float*)d_in[0];   // [N, IN_DIM]  f32
  const int*   xtype  = (const int*)d_in[1];     // [N]          i32
  const float* weight = (const float*)d_in[2];   // [NUM_B, IN_DIM*OUT_DIM] f32
  const float* w_comp = (const float*)d_in[3];   // [NUM_T, NUM_B] f32
  float*       out    = (float*)d_out;           // [N, OUT_DIM] f32

  // workspace: Wt (8 MB f16, transposed per-type weights) then x16 (2 MB f16)
  _Float16* Wt  = (_Float16*)d_ws;
  _Float16* x16 = (_Float16*)((char*)d_ws + (size_t)NUM_T * IN_DIM * OUT_DIM * sizeof(_Float16));

  build_wt<<<dim3(NUM_T / 4, 4), 256, 0, stream>>>(weight, w_comp, Wt);
  cvt_x<<<(NN * IN_DIM) / 256, 256, 0, stream>>>(x, x16);
  rgcn_gemm<<<dim3(NUM_T, 4), 256, 0, stream>>>(x16, xtype, Wt, out);
}
